// MultiHeadSelfAttention_11158325035343
// MI455X (gfx1250) — matrix operand order repaired and checked by reading, not compiled, not run on hardware
//
#include <hip/hip_runtime.h>

// ---------------------------------------------------------------------------
// Problem constants (from reference)
// ---------------------------------------------------------------------------
constexpr int B_SZ  = 4;
constexpr int T_SZ  = 2048;
constexpr int E_SZ  = 2048;
constexpr int H_Q   = 16;
constexpr int H_KV  = 8;
constexpr int HD_SZ = 128;
constexpr int QN    = H_Q  * HD_SZ;   // 2048
constexpr int KN    = H_KV * HD_SZ;   // 1024
constexpr int M_SZ  = B_SZ * T_SZ;    // 8192 rows

typedef _Float16 half_t;
typedef __attribute__((ext_vector_type(16))) _Float16 v16h;
typedef __attribute__((ext_vector_type(8)))  float    v8f;

// CDNA5 async global->LDS copy (ASYNCcnt path), with safe fallback.
#if defined(__has_builtin)
#if __has_builtin(__builtin_amdgcn_global_load_async_to_lds_b128) && \
    __has_builtin(__builtin_amdgcn_s_wait_asynccnt)
#define USE_ASYNC_LDS 1
#endif
#endif

// Builtin parameter type on this toolchain: int vector_size(16), AS1/AS3 ptrs.
typedef int b128_t __attribute__((vector_size(16)));
typedef __attribute__((address_space(1))) b128_t* gptr_b128;
typedef __attribute__((address_space(3))) b128_t* lptr_b128;

__device__ __forceinline__ void copy16B(const half_t* __restrict__ g, half_t* l) {
#ifdef USE_ASYNC_LDS
  __builtin_amdgcn_global_load_async_to_lds_b128((gptr_b128)g, (lptr_b128)l, 0, 0);
#else
  *(uint4*)l = *(const uint4*)g;
#endif
}

__device__ __forceinline__ void wait_async_copies() {
#ifdef USE_ASYNC_LDS
  __builtin_amdgcn_s_wait_asynccnt(0);
#endif
}

// ---------------------------------------------------------------------------
// WMMA fragment loaders (CDNA5 ISA 7.12.2 layouts, wave32)
// ---------------------------------------------------------------------------
// A: 16x32 f16, row-major source with leading dim lda.
// lanes 0-15: row = lane, K = {0..7,16..23}; lanes 16-31: row=lane-16, K={8..15,24..31}
__device__ __forceinline__ v16h load_a_frag(const half_t* __restrict__ p, int lda) {
  const int lane = threadIdx.x & 31;
  const int row  = lane & 15;
  const int kb   = (lane >> 4) << 3;           // 0 or 8
  const half_t* r = p + (size_t)row * lda;
  v16h a;
#pragma unroll
  for (int i = 0; i < 8; ++i) a[i] = r[kb + i];
#pragma unroll
  for (int i = 0; i < 8; ++i) a[8 + i] = r[16 + kb + i];
  return a;
}

// Logical B[k][n] = p[n*ld + k] (transpose of a row-major matrix).
// lanes 0-15: col = lane, K = {0..7,16..23} ; lanes 16-31: col = lane-16, K = {8..15,24..31}
__device__ __forceinline__ v16h load_bT_frag(const half_t* __restrict__ p, int ld) {
  const int lane = threadIdx.x & 31;
  const int col  = lane & 15;
  const int kb   = (lane >> 4) << 3;           // 0 or 8
  const half_t* r = p + (size_t)col * ld;
  v16h bf;
#pragma unroll
  for (int i = 0; i < 8; ++i) bf[i] = r[kb + i];
#pragma unroll
  for (int i = 0; i < 8; ++i) bf[8 + i] = r[16 + kb + i];
  return bf;
}

// ---------------------------------------------------------------------------
// Casts: plain f32->f16, and f32->f16 with transpose (for weights)
// ---------------------------------------------------------------------------
__global__ void cast_kernel(const float* __restrict__ src, half_t* __restrict__ dst, int n) {
  int i = blockIdx.x * blockDim.x + threadIdx.x;
  if (i < n) dst[i] = (half_t)src[i];
}

// src: [R][Ncols] f32 (row-major). dst: [Ncols][R] f16 (row-major).
__global__ void cast_transpose_kernel(const float* __restrict__ src, half_t* __restrict__ dst,
                                      int R, int Ncols) {
  int idx = blockIdx.x * blockDim.x + threadIdx.x;
  if (idx >= R * Ncols) return;
  const int r = idx % R;
  const int c = idx / R;
  dst[(size_t)c * R + r] = (half_t)src[(size_t)r * Ncols + c];
}

// ---------------------------------------------------------------------------
// LDS-tiled WMMA GEMM: C[M,N] = A[M,K] * Bt[N,K]^T (both f16), f32 accumulate.
// Block = 256 threads (8 waves). Block tile 64(M) x 128(N), K step 32.
// Wave (wy 0..3, wx 0..1) owns a 16x64 strip: 1 A-frag shared by 4 WMMAs.
// Tiles staged in LDS with padded stride (40 halfs = 80B, 16B aligned).
// VTOUT: scatter output as V^T [b][n][t] for the attention PV path.
// ---------------------------------------------------------------------------
template <typename OutT, bool VTOUT>
__global__ void __launch_bounds__(256)
wmma_gemm_tiled(const half_t* __restrict__ A, const half_t* __restrict__ Bt,
                OutT* __restrict__ C, int M, int N, int K) {
  constexpr int LDSP = 40;                         // padded row stride (halfs)
  __shared__ half_t As[64 * LDSP];                 // 5.0 KB
  __shared__ half_t Bs[128 * LDSP];                // 10.0 KB

  const int tid  = threadIdx.x;
  const int wave = tid >> 5;
  const int wy   = wave >> 1;                      // 0..3  (M)
  const int wx   = wave & 1;                       // 0..1  (N)
  const int m0   = blockIdx.y * 64;
  const int n0   = blockIdx.x * 128;

  const int crow = tid >> 2;                       // 0..63
  const int cchk = (tid & 3) * 8;                  // 0,8,16,24 (halfs; 16B chunks)

  v8f acc[4];
  const v8f vz = {};
#pragma unroll
  for (int nn = 0; nn < 4; ++nn) acc[nn] = vz;

  for (int k0 = 0; k0 < K; k0 += 32) {
    // stage A (64x32) and Bt (128x32) tiles
    copy16B(A  + (size_t)(m0 + crow) * K + k0 + cchk, As + crow * LDSP + cchk);
    copy16B(Bt + (size_t)(n0 + crow) * K + k0 + cchk, Bs + crow * LDSP + cchk);
    copy16B(Bt + (size_t)(n0 + crow + 64) * K + k0 + cchk, Bs + (crow + 64) * LDSP + cchk);
    wait_async_copies();
    __syncthreads();

    v16h af = load_a_frag(As + wy * 16 * LDSP, LDSP);
#pragma unroll
    for (int nn = 0; nn < 4; ++nn) {
      v16h bf = load_bT_frag(Bs + (wx * 64 + nn * 16) * LDSP, LDSP);
      acc[nn] = __builtin_amdgcn_wmma_f32_16x16x32_f16(false, af, false, bf,
                                                       (short)0, acc[nn], false, false);
    }
    __syncthreads();
  }

  const int lane = tid & 31;
  const int col  = lane & 15;
  const int mb   = (lane >> 4) << 3;
#pragma unroll
  for (int nn = 0; nn < 4; ++nn) {
    const int n = n0 + wx * 64 + nn * 16 + col;
#pragma unroll
    for (int j = 0; j < 8; ++j) {
      const int m = m0 + wy * 16 + mb + j;
      if constexpr (VTOUT) {
        const int bb = m / T_SZ;
        const int t  = m % T_SZ;
        C[((size_t)bb * KN + n) * T_SZ + t] = (OutT)acc[nn][j];
      } else {
        C[(size_t)m * N + n] = (OutT)acc[nn][j];
      }
    }
  }
}

// ---------------------------------------------------------------------------
// xPos rotary, in place on f16 [.., t, nheads, 128]. One thread per (row, d<64)
// pair. invert!=0 divides by the xPos scale (K path).
// ---------------------------------------------------------------------------
__global__ void rope_kernel(half_t* __restrict__ qk, int nheads, int total, int invert) {
  int idx = blockIdx.x * blockDim.x + threadIdx.x;
  if (idx >= total) return;
  const int d  = idx & 63;
  const int th = idx >> 6;                 // (b*T + t)*nheads + h
  const int t  = (th / nheads) % T_SZ;

  const size_t base = (size_t)th * HD_SZ + d;

  const float inv_freq = __expf(-((float)(2 * d) * (1.0f / 128.0f)) * 9.210340371976184f);
  const float ang = (float)t * inv_freq;
  float sn, cs;
  __sincosf(ang, &sn, &cs);

  const float sv = ((float)(2 * d) + 51.2f) * (1.0f / 179.2f);
  float power = ((float)t - (float)(T_SZ / 2)) * (1.0f / 512.0f);
  if (invert) power = -power;
  const float sc = __expf(power * __logf(sv));

  const float x1 = (float)qk[base]      * sc;
  const float x2 = (float)qk[base + 64] * sc;
  qk[base]      = (half_t)(x1 * cs - x2 * sn);
  qk[base + 64] = (half_t)(x2 * cs + x1 * sn);
}

// ---------------------------------------------------------------------------
// Causal flash attention. One wave per (b, h, 16-query tile).
// S = Q K^T via WMMA; online softmax with cross-lane reductions matched to the
// C layout (lane = column, VGPR j = row); P staged through wave-private LDS;
// P*V uses the transposed V buffer so B fragments are lane-contiguous.
// ---------------------------------------------------------------------------
__global__ void __launch_bounds__(128)
attn_kernel(const half_t* __restrict__ Q, const half_t* __restrict__ Kh,
            const half_t* __restrict__ Vt, half_t* __restrict__ O) {
  __shared__ half_t plds[4][16 * 32];

  const int wave = threadIdx.x >> 5;
  const int lane = threadIdx.x & 31;
  const int wid  = blockIdx.x * 4 + wave;

  const int nqt = T_SZ / 16;
  const int qt  = wid % nqt;
  const int h   = (wid / nqt) % H_Q;
  const int b   = wid / (nqt * H_Q);
  const int kvh = h / (H_Q / H_KV);
  const int q0  = qt * 16;

  const int ldq = H_Q  * HD_SZ;
  const int ldk = H_KV * HD_SZ;
  const half_t* qbase  = Q  + ((size_t)(b * T_SZ + q0) * H_Q + h) * HD_SZ;
  const half_t* kbase  = Kh + ((size_t)b * T_SZ * H_KV + kvh) * HD_SZ;
  const half_t* vtbase = Vt + ((size_t)b * KN + kvh * HD_SZ) * T_SZ;  // [d][t]

  v16h aq[4];
#pragma unroll
  for (int c = 0; c < 4; ++c) aq[c] = load_a_frag(qbase + 32 * c, ldq);

  v8f accv[8];
  const v8f vzero = {};
#pragma unroll
  for (int d = 0; d < 8; ++d) accv[d] = vzero;

  float mrow[8], lrow[8];
#pragma unroll
  for (int j = 0; j < 8; ++j) { mrow[j] = -__builtin_inff(); lrow[j] = 0.0f; }

  const float inv_sqrt_d = 0.08838834764831845f;
  const int col    = lane & 15;
  const int rowoff = (lane >> 4) << 3;
  const int qhi    = q0 + 15;

  half_t* pst = plds[wave];

  for (int tk0 = 0; tk0 <= qhi; tk0 += 32) {
    float s[2][8];
#pragma unroll
    for (int half2 = 0; half2 < 2; ++half2) {
      const int c0 = tk0 + 16 * half2;
      v8f sacc = vzero;
#pragma unroll
      for (int c = 0; c < 4; ++c) {
        v16h bk = load_bT_frag(kbase + (size_t)c0 * ldk + 32 * c, ldk);
        sacc = __builtin_amdgcn_wmma_f32_16x16x32_f16(false, aq[c], false, bk,
                                                      (short)0, sacc, false, false);
      }
#pragma unroll
      for (int j = 0; j < 8; ++j) {
        const int krow = c0 + col;
        const int qrow = q0 + rowoff + j;
        s[half2][j] = (krow > qrow) ? -__builtin_inff() : sacc[j] * inv_sqrt_d;
      }
    }

#pragma unroll
    for (int j = 0; j < 8; ++j) {
      float v = fmaxf(s[0][j], s[1][j]);
      v = fmaxf(v, __shfl_xor(v, 1, 32));
      v = fmaxf(v, __shfl_xor(v, 2, 32));
      v = fmaxf(v, __shfl_xor(v, 4, 32));
      v = fmaxf(v, __shfl_xor(v, 8, 32));
      const float nm = fmaxf(mrow[j], v);

      const float p0 = __expf(s[0][j] - nm);
      const float p1 = __expf(s[1][j] - nm);
      float rs = p0 + p1;
      rs += __shfl_xor(rs, 1, 32);
      rs += __shfl_xor(rs, 2, 32);
      rs += __shfl_xor(rs, 4, 32);
      rs += __shfl_xor(rs, 8, 32);

      const float alpha = __expf(mrow[j] - nm);
      lrow[j] = lrow[j] * alpha + rs;
      mrow[j] = nm;
#pragma unroll
      for (int d = 0; d < 8; ++d) accv[d][j] *= alpha;

      pst[(rowoff + j) * 32 + col]      = (half_t)p0;
      pst[(rowoff + j) * 32 + 16 + col] = (half_t)p1;
    }

    v16h ap = load_a_frag(pst, 32);
#pragma unroll
    for (int d = 0; d < 8; ++d) {
      // logical B[k=token][n=d]: V^T row (d) is token-contiguous
      v16h bv = load_bT_frag(vtbase + (size_t)(d * 16) * T_SZ + tk0, T_SZ);
      accv[d] = __builtin_amdgcn_wmma_f32_16x16x32_f16(false, ap, false, bv,
                                                       (short)0, accv[d], false, false);
    }
  }

  half_t* obase = O + ((size_t)b * T_SZ * H_Q + h) * HD_SZ;
#pragma unroll
  for (int d = 0; d < 8; ++d) {
#pragma unroll
    for (int j = 0; j < 8; ++j) {
      const int qrow = q0 + rowoff + j;
      obase[(size_t)qrow * ldq + d * 16 + col] = (half_t)(accv[d][j] / lrow[j]);
    }
  }
}

// ---------------------------------------------------------------------------
// Host-side orchestration
// ---------------------------------------------------------------------------
extern "C" void kernel_launch(void* const* d_in, const int* in_sizes, int n_in,
                              void* d_out, int out_size, void* d_ws, size_t ws_size,
                              hipStream_t stream) {
  const float* x  = (const float*)d_in[0];
  const float* Wq = (const float*)d_in[1];
  const float* Wk = (const float*)d_in[2];
  const float* Wv = (const float*)d_in[3];
  const float* Wo = (const float*)d_in[4];

  half_t* w = (half_t*)d_ws;
  size_t o = 0;
  half_t* xh  = w + o; o += (size_t)M_SZ * E_SZ;     // x,  [M][E]
  half_t* wqt = w + o; o += (size_t)E_SZ * QN;       // Wq^T [QN][E]
  half_t* wkt = w + o; o += (size_t)E_SZ * KN;       // Wk^T [KN][E]
  half_t* wvt = w + o; o += (size_t)E_SZ * KN;       // Wv^T [KN][E]
  half_t* wot = w + o; o += (size_t)E_SZ * E_SZ;     // Wo^T [E][E]
  half_t* qh  = w + o; o += (size_t)M_SZ * QN;       // Q   [M][QN]
  half_t* kh  = w + o; o += (size_t)M_SZ * KN;       // K   [M][KN]
  half_t* vt  = w + o; o += (size_t)M_SZ * KN;       // V^T [B][KN][T]
  half_t* ah  = w + o; o += (size_t)M_SZ * E_SZ;     // attn out [M][E]

  cast_kernel<<<(M_SZ * E_SZ) / 256, 256, 0, stream>>>(x, xh, M_SZ * E_SZ);
  cast_transpose_kernel<<<(E_SZ * QN) / 256, 256, 0, stream>>>(Wq, wqt, E_SZ, QN);
  cast_transpose_kernel<<<(E_SZ * KN) / 256, 256, 0, stream>>>(Wk, wkt, E_SZ, KN);
  cast_transpose_kernel<<<(E_SZ * KN) / 256, 256, 0, stream>>>(Wv, wvt, E_SZ, KN);
  cast_transpose_kernel<<<(E_SZ * E_SZ) / 256, 256, 0, stream>>>(Wo, wot, E_SZ, E_SZ);

  dim3 blk(256);
  // QKV projections (V written transposed for the attention PV path)
  wmma_gemm_tiled<half_t, false><<<dim3(QN / 128, M_SZ / 64), blk, 0, stream>>>(
      xh, wqt, qh, M_SZ, QN, E_SZ);
  wmma_gemm_tiled<half_t, false><<<dim3(KN / 128, M_SZ / 64), blk, 0, stream>>>(
      xh, wkt, kh, M_SZ, KN, E_SZ);
  wmma_gemm_tiled<half_t, true><<<dim3(KN / 128, M_SZ / 64), blk, 0, stream>>>(
      xh, wvt, vt, M_SZ, KN, E_SZ);

  // xPos rotary (Q: *scale, K: /scale). V^T needs no rotary.
  {
    int totq = M_SZ * H_Q  * (HD_SZ / 2);
    int totk = M_SZ * H_KV * (HD_SZ / 2);
    rope_kernel<<<totq / 256, 256, 0, stream>>>(qh, H_Q,  totq, 0);
    rope_kernel<<<totk / 256, 256, 0, stream>>>(kh, H_KV, totk, 1);
  }

  // causal flash attention
  {
    int nwaves = B_SZ * H_Q * (T_SZ / 16);   // 8192
    attn_kernel<<<nwaves / 4, 128, 0, stream>>>(qh, kh, vt, ah);
  }

  // output projection -> f32 d_out
  wmma_gemm_tiled<float, false><<<dim3(E_SZ / 128, M_SZ / 64), blk, 0, stream>>>(
      ah, wot, (float*)d_out, M_SZ, E_SZ, E_SZ);
}
